// WindowAttention_48524540510914
// MI455X (gfx1250) — matrix operand order repaired and checked by reading, not compiled, not run on hardware
//
#include <hip/hip_runtime.h>
#include <hip/hip_bf16.h>
#include <math.h>

typedef __attribute__((ext_vector_type(16))) _Float16 v16h;
typedef __attribute__((ext_vector_type(8)))  float    v8f;

#define DEV __device__ __forceinline__

// ---- problem dims -----------------------------------------------------------
static constexpr int BQ   = 512;          // windows
static constexpr int NT   = 49;           // tokens per window (7x7)
static constexpr int CD   = 256;          // channels
static constexpr int HH   = 8;            // heads
static constexpr int HD   = 32;           // head dim
static constexpr int RB   = 8;            // ref batch
static constexpr int NRF  = 64;           // ref tokens
static constexpr int NWIN = 64;           // BQ / RB
static constexpr int SPAT = NWIN * NT;    // 3136  (conv H)
static constexpr int CONVW = 64;          // conv W (= NRF)
static constexpr float SCALE  = 0.17677669529663687f;   // 32^-0.5
static constexpr float LNEPS  = 1e-5f;
static constexpr int OUTBASE = BQ * NT * CD;            // 6422528

// ---- WMMA helpers (CDNA5 v_wmma_f32_16x16x32_f16) --------------------------
DEV v8f wmma16(v16h a, v16h b, v8f c) {
  return __builtin_amdgcn_wmma_f32_16x16x32_f16(false, a, false, b, (short)0, c,
                                                false, false);
}

// A fragment (16x32, M x K) from fp32 row-major A, leading dim lda.
DEV v16h loadA_f32(const float* __restrict__ A, int lda, int m0, int kbase) {
  int lane = threadIdx.x & 31;
  int m = m0 + (lane & 15);
  int koff = (lane >> 4) << 4;
  const float* p = A + m * lda + kbase + koff;
  v16h a;
#pragma unroll
  for (int i = 0; i < 8; ++i) {
    int k = 2 * i;
    a[2 * i]     = (_Float16)p[k];
    a[2 * i + 1] = (_Float16)p[k + 1];
  }
  return a;
}

// A fragment from f16 row-major A, row index clamped to mClamp.
DEV v16h loadA_f16(const _Float16* __restrict__ A, int lda, int m0, int kbase,
                   int mClamp) {
  int lane = threadIdx.x & 31;
  int m = m0 + (lane & 15);
  if (m > mClamp) m = mClamp;
  int koff = (lane >> 4) << 4;
  const _Float16* p = A + m * lda + kbase + koff;
  v16h a;
#pragma unroll
  for (int i = 0; i < 8; ++i) {
    int k = 2 * i;
    a[2 * i]     = p[k];
    a[2 * i + 1] = p[k + 1];
  }
  return a;
}

// B fragment (32x16, K x N) from fp32 row-major B[k][n], leading dim ldb.
DEV v16h loadB_f32(const float* __restrict__ B, int ldb, int n0, int kbase) {
  int lane = threadIdx.x & 31;
  int n = n0 + (lane & 15);
  int kb = kbase + ((lane >> 4) << 4);
  v16h b;
#pragma unroll
  for (int j = 0; j < 16; ++j) b[j] = (_Float16)B[(kb + j) * ldb + n];
  return b;
}

// B fragment from f16 row-major B[k][n], k clamped to kClamp.
DEV v16h loadB_f16(const _Float16* __restrict__ B, int ldb, int n0, int kbase,
                   int kClamp) {
  int lane = threadIdx.x & 31;
  int n = n0 + (lane & 15);
  int kb = kbase + ((lane >> 4) << 4);
  v16h b;
#pragma unroll
  for (int j = 0; j < 16; ++j) {
    int k = kb + j;
    if (k > kClamp) k = kClamp;
    b[j] = B[k * ldb + n];
  }
  return b;
}

// B fragment where B = M^T and M is f16 row-major (rows = n, cols = k).
DEV v16h loadBT_f16(const _Float16* __restrict__ M, int ldm, int n0, int kbase,
                    int nClamp) {
  int lane = threadIdx.x & 31;
  int n = n0 + (lane & 15);
  if (n > nClamp) n = nClamp;
  int kb = kbase + ((lane >> 4) << 4);
  const _Float16* p = M + n * ldm + kb;
  v16h b;
#pragma unroll
  for (int j = 0; j < 16; ++j) b[j] = p[j];
  return b;
}

// ---- K1: qkv = x @ qkv_w + b ; split into q(*scale)/k/v f16 (b,h,n,d) ------
__global__ void k_qkv(const float* __restrict__ x, const float* __restrict__ w,
                      const float* __restrict__ bias, _Float16* __restrict__ q16,
                      _Float16* __restrict__ k16, _Float16* __restrict__ vh16) {
  int wid = blockIdx.x * (blockDim.x >> 5) + (threadIdx.x >> 5);
  const int TN = (3 * CD) / 16;  // 48
  int tm = wid / TN, tn = wid % TN;
  v8f c = {};
  for (int ks = 0; ks < CD; ks += 32) {
    v16h a = loadA_f32(x, CD, tm * 16, ks);
    v16h b = loadB_f32(w, 3 * CD, tn * 16, ks);
    c = wmma16(a, b, c);
  }
  int lane = threadIdx.x & 31;
  int gn = tn * 16 + (lane & 15);
  float bv = bias[gn];
  int sel = gn >> 8;       // 0:q 1:k 2:v
  int cc = gn & 255;
  int h = cc >> 5, d = cc & 31;
  _Float16* dst = (sel == 0) ? q16 : ((sel == 1) ? k16 : vh16);
  float mult = (sel == 0) ? SCALE : 1.0f;
#pragma unroll
  for (int r = 0; r < 8; ++r) {
    int gm = tm * 16 + r + ((lane >> 4) << 3);
    int bi = gm / NT, tok = gm % NT;
    dst[((bi * HH + h) * NT + tok) * HD + d] = (_Float16)((c[r] + bv) * mult);
  }
}

// ---- K2: ref qk GEMM + diff transform --------------------------------------
__global__ void k_refqk(const float* __restrict__ xr, const float* __restrict__ w,
                        const float* __restrict__ bias,
                        const float* __restrict__ dmu,
                        const float* __restrict__ dls,
                        _Float16* __restrict__ rq16, _Float16* __restrict__ rv16) {
  int wid = blockIdx.x * (blockDim.x >> 5) + (threadIdx.x >> 5);
  const int TN = (2 * CD) / 16;  // 32
  int tm = wid / TN, tn = wid % TN;
  v8f c = {};
  for (int ks = 0; ks < CD; ks += 32) {
    v16h a = loadA_f32(xr, CD, tm * 16, ks);
    v16h b = loadB_f32(w, 2 * CD, tn * 16, ks);
    c = wmma16(a, b, c);
  }
  int lane = threadIdx.x & 31;
  int gn = tn * 16 + (lane & 15);
  float bv = bias[gn];
  int sel = gn >> 8;  // 0: ref_q, 1: ref_v
  int cc = gn & 255;
  int h = cc >> 5, d = cc & 31;
  float mu = dmu[cc], sg = expf(dls[cc]);
  _Float16* dst = (sel == 0) ? rq16 : rv16;
#pragma unroll
  for (int r = 0; r < 8; ++r) {
    int gm = tm * 16 + r + ((lane >> 4) << 3);
    int rb = gm >> 6, m = gm & 63;
    float val = c[r] + bv;
    if (sel == 0) val = mu + sg * val;
    dst[((rb * HH + h) * NRF + m) * HD + d] = (_Float16)val;
  }
}

// ---- K3: ref_attn = (q*scale) @ ref_k^T -> r buffer (rb,h,w*49+tok,m) ------
__global__ void k_refattn(const _Float16* __restrict__ q16,
                          const _Float16* __restrict__ rq16,
                          float* __restrict__ rbuf) {
  int wid = blockIdx.x * (blockDim.x >> 5) + (threadIdx.x >> 5);
  int t = wid & 15;
  int bh = wid >> 4;
  int b = bh >> 3, h = bh & 7;
  int tm = t >> 2, tn = t & 3;
  int rb = b >> 6, wwin = b & 63;
  const _Float16* A = q16 + (b * HH + h) * NT * HD;
  const _Float16* K = rq16 + (rb * HH + h) * NRF * HD;
  v16h a = loadA_f16(A, HD, tm * 16, 0, NT - 1);
  v16h bf = loadBT_f16(K, HD, tn * 16, 0, NRF - 1);
  v8f c = {};
  c = wmma16(a, bf, c);
  int lane = threadIdx.x & 31;
  int n = tn * 16 + (lane & 15);
  float* dst = rbuf + ((rb * HH + h) * SPAT + wwin * NT) * NRF;
#pragma unroll
  for (int r = 0; r < 8; ++r) {
    int m = tm * 16 + r + ((lane >> 4) << 3);
    if (m < NT) dst[m * NRF + n] = c[r];
  }
}

// ---- K4: 3x3 SAME conv over (rb, 8ch, 3136, 64) ----------------------------
__global__ void k_conv(const float* __restrict__ rbuf,
                       const float* __restrict__ cw, const float* __restrict__ cb,
                       float* __restrict__ ubuf) {
  __shared__ float wsh[HH * HH * 9];
  __shared__ float bsh[HH];
  for (int i = threadIdx.x; i < HH * HH * 9; i += blockDim.x) wsh[i] = cw[i];
  if (threadIdx.x < HH) bsh[threadIdx.x] = cb[threadIdx.x];
  __syncthreads();
  int idx = blockIdx.x * blockDim.x + threadIdx.x;  // < 12,845,056
  int xw = idx & 63;
  int rest = idx >> 6;
  int y = rest % SPAT;
  rest /= SPAT;
  int ho = rest & 7, rb = rest >> 3;
  const float* in = rbuf + rb * HH * SPAT * CONVW;
  float acc = bsh[ho];
  for (int hi = 0; hi < HH; ++hi) {
    const float* inp = in + hi * SPAT * CONVW;
    const float* wp = wsh + (ho * HH + hi) * 9;
#pragma unroll
    for (int dy = -1; dy <= 1; ++dy) {
      int yy = y + dy;
      if (yy < 0 || yy >= SPAT) continue;
#pragma unroll
      for (int dx = -1; dx <= 1; ++dx) {
        int xx = xw + dx;
        if (xx < 0 || xx >= CONVW) continue;
        acc += wp[(dy + 1) * 3 + (dx + 1)] * inp[yy * CONVW + xx];
      }
    }
  }
  ubuf[idx] = acc;
}

// ---- K5: per-(rb,h) mean/var over 3136x64 ----------------------------------
__global__ void k_stats(const float* __restrict__ ubuf, float* __restrict__ stats) {
  int ch = blockIdx.x;  // 0..63
  const float* p = ubuf + ch * SPAT * CONVW;
  float s = 0.f, s2 = 0.f;
  for (int i = threadIdx.x; i < SPAT * CONVW; i += blockDim.x) {
    float v = p[i];
    s += v;
    s2 += v * v;
  }
  __shared__ float sh[256], sh2[256];
  int tid = threadIdx.x;
  sh[tid] = s;
  sh2[tid] = s2;
  __syncthreads();
  for (int o = 128; o > 0; o >>= 1) {
    if (tid < o) {
      sh[tid] += sh[tid + o];
      sh2[tid] += sh2[tid + o];
    }
    __syncthreads();
  }
  if (tid == 0) {
    const float inv = 1.0f / (float)(SPAT * CONVW);
    float mu = sh[0] * inv;
    float var = sh2[0] * inv - mu * mu;
    stats[ch * 2] = mu;
    stats[ch * 2 + 1] = var;
  }
}

// ---- K6: r += gelu_exact((u - mu) * rsqrt(var + eps)) ----------------------
__global__ void k_gelu(float* __restrict__ rbuf, const float* __restrict__ ubuf,
                       const float* __restrict__ stats) {
  int idx = blockIdx.x * blockDim.x + threadIdx.x;
  int ch = idx / (SPAT * CONVW);
  float mu = stats[ch * 2], var = stats[ch * 2 + 1];
  float xn = (ubuf[idx] - mu) * rsqrtf(var + LNEPS);
  float g = 0.5f * xn * (1.0f + erff(xn * 0.70710678118654752f));
  rbuf[idx] += g;
}

// ---- K7: fused softmax(64) + q_new = probs @ ref_v * scale -----------------
// Score tile is staged global->LDS with async-to-LDS DMA (ASYNCcnt path).
__global__ void k_refsm_qnew(const float* __restrict__ rbuf,
                             const _Float16* __restrict__ rv16,
                             _Float16* __restrict__ qn16) {
  int bh = blockIdx.x;
  int b = bh >> 3, h = bh & 7;
  int rb = b >> 6, wwin = b & 63;
  __shared__ float __attribute__((aligned(16))) sc[NT * NRF];
  __shared__ _Float16 pr[64 * NRF];
  const float* src = rbuf + ((rb * HH + h) * SPAT + wwin * NT) * NRF;
  int tid = threadIdx.x;

  // async global -> LDS: 49*64 floats = 784 x 16B chunks (src 256B aligned)
  {
    unsigned lds_base = (unsigned)(size_t)(&sc[0]);  // low 32 bits = LDS offset
    unsigned long long gbase = (unsigned long long)(size_t)src;
    for (int chnk = tid; chnk < (NT * NRF) / 4; chnk += 256) {
      unsigned lofs = lds_base + (unsigned)chnk * 16u;
      unsigned long long ga = gbase + (unsigned long long)chnk * 16ull;
      asm volatile("global_load_async_to_lds_b128 %0, %1, off"
                   :: "v"(lofs), "v"(ga) : "memory");
    }
    asm volatile("s_wait_asynccnt 0" ::: "memory");
  }
  __syncthreads();

  if (tid < 64) {
    if (tid < NT) {
      float mx = -1e30f;
      for (int m = 0; m < NRF; ++m) mx = fmaxf(mx, sc[tid * NRF + m]);
      float s = 0.f;
      for (int m = 0; m < NRF; ++m) {
        float e = expf(sc[tid * NRF + m] - mx);
        s += e;
        sc[tid * NRF + m] = e;
      }
      float inv = 1.0f / s;
      for (int m = 0; m < NRF; ++m)
        pr[tid * NRF + m] = (_Float16)(sc[tid * NRF + m] * inv);
    } else {
      for (int m = 0; m < NRF; ++m) pr[tid * NRF + m] = (_Float16)0.0f;
    }
  }
  __syncthreads();
  int wv = tid >> 5, lane = tid & 31;
  int tm = wv >> 1, tn = wv & 1;  // 4 M-tiles x 2 N-tiles
  const _Float16* Bv = rv16 + (rb * HH + h) * NRF * HD;
  v8f c = {};
  for (int ks = 0; ks < NRF; ks += 32) {
    v16h a = loadA_f16(pr, NRF, tm * 16, ks, 63);
    v16h bf = loadB_f16(Bv, HD, tn * 16, ks, NRF - 1);
    c = wmma16(a, bf, c);
  }
  int n = tn * 16 + (lane & 15);
  _Float16* dst = qn16 + (b * HH + h) * NT * HD;
#pragma unroll
  for (int r = 0; r < 8; ++r) {
    int m = tm * 16 + r + ((lane >> 4) << 3);
    if (m < NT) dst[m * HD + n] = (_Float16)(c[r] * SCALE);
  }
}

// ---- K8: fused window attn: scores(+rpb+mask) -> softmax(49) -> PV ---------
__global__ void k_winattn(const _Float16* __restrict__ qn16,
                          const _Float16* __restrict__ k16,
                          const _Float16* __restrict__ vh16,
                          const float* __restrict__ rpb_table,
                          const float* __restrict__ mask,
                          _Float16* __restrict__ outpre) {
  int bh = blockIdx.x;
  int b = bh >> 3, h = bh & 7;
  __shared__ float sc[NT * NT];
  __shared__ _Float16 pr[64 * 64];
  const _Float16* Aq = qn16 + (b * HH + h) * NT * HD;
  const _Float16* Kk = k16 + (b * HH + h) * NT * HD;
  int tid = threadIdx.x, wv = tid >> 5, lane = tid & 31;
  for (int t = wv; t < 16; t += 8) {
    int tm = t >> 2, tn = t & 3;
    v16h a = loadA_f16(Aq, HD, tm * 16, 0, NT - 1);
    v16h bf = loadBT_f16(Kk, HD, tn * 16, 0, NT - 1);
    v8f c = {};
    c = wmma16(a, bf, c);
    int n = tn * 16 + (lane & 15);
    if (n < NT) {
      int jy = n / 7, jx = n % 7;
#pragma unroll
      for (int r = 0; r < 8; ++r) {
        int m = tm * 16 + r + ((lane >> 4) << 3);
        if (m < NT) {
          int iy = m / 7, ix = m % 7;
          int idx = (iy - jy + 6) * 13 + (ix - jx + 6);
          float bias = rpb_table[idx * HH + h] +
                       mask[((b & 63) * NT + m) * NT + n];
          sc[m * NT + n] = c[r] + bias;
        }
      }
    }
  }
  __syncthreads();
  if (tid < 64) {
    if (tid < NT) {
      float mx = -1e30f;
      for (int n = 0; n < NT; ++n) mx = fmaxf(mx, sc[tid * NT + n]);
      float s = 0.f;
      for (int n = 0; n < NT; ++n) {
        float e = expf(sc[tid * NT + n] - mx);
        s += e;
        sc[tid * NT + n] = e;
      }
      float inv = 1.0f / s;
      for (int n = 0; n < NT; ++n)
        pr[tid * 64 + n] = (_Float16)(sc[tid * NT + n] * inv);
      for (int n = NT; n < 64; ++n) pr[tid * 64 + n] = (_Float16)0.0f;
    } else {
      for (int n = 0; n < 64; ++n) pr[tid * 64 + n] = (_Float16)0.0f;
    }
  }
  __syncthreads();
  int tm = wv >> 1, tn = wv & 1;
  const _Float16* Bv = vh16 + (b * HH + h) * NT * HD;
  v8f c = {};
  for (int ks = 0; ks < 64; ks += 32) {
    v16h a = loadA_f16(pr, 64, tm * 16, ks, 63);
    v16h bf = loadB_f16(Bv, HD, tn * 16, ks, NT - 1);  // k clamped; probs 0 there
    c = wmma16(a, bf, c);
  }
  int n = tn * 16 + (lane & 15);
  _Float16* dst = outpre + b * NT * CD;
#pragma unroll
  for (int r = 0; r < 8; ++r) {
    int m = tm * 16 + r + ((lane >> 4) << 3);
    if (m < NT) dst[m * CD + h * HD + n] = (_Float16)c[r];
  }
}

// ---- K9: out = outpre @ proj_w + proj_b ------------------------------------
__global__ void k_proj(const _Float16* __restrict__ outpre,
                       const float* __restrict__ w, const float* __restrict__ bias,
                       float* __restrict__ out) {
  int wid = blockIdx.x * (blockDim.x >> 5) + (threadIdx.x >> 5);
  const int TN = CD / 16;  // 16
  int tm = wid / TN, tn = wid % TN;
  v8f c = {};
  for (int ks = 0; ks < CD; ks += 32) {
    v16h a = loadA_f16(outpre, CD, tm * 16, ks, BQ * NT - 1);
    v16h b = loadB_f32(w, CD, tn * 16, ks);
    c = wmma16(a, b, c);
  }
  int lane = threadIdx.x & 31;
  int gn = tn * 16 + (lane & 15);
  float bv = bias[gn];
#pragma unroll
  for (int r = 0; r < 8; ++r) {
    int gm = tm * 16 + r + ((lane >> 4) << 3);
    out[gm * CD + gn] = c[r] + bv;
  }
}

// ---- K10: pass-through tokens ----------------------------------------------
__global__ void k_tokens(const float* __restrict__ dt, const float* __restrict__ st,
                         float* __restrict__ out) {
  int i = blockIdx.x * blockDim.x + threadIdx.x;
  if (i < RB * CD) out[OUTBASE + i] = dt[i];
  else if (i < 2 * RB * CD) out[OUTBASE + i] = st[i - RB * CD];
}

// ---- host launcher ----------------------------------------------------------
extern "C" void kernel_launch(void* const* d_in, const int* in_sizes, int n_in,
                              void* d_out, int out_size, void* d_ws, size_t ws_size,
                              hipStream_t stream) {
  (void)in_sizes; (void)n_in; (void)out_size; (void)ws_size;
  const float* x     = (const float*)d_in[0];
  const float* mask  = (const float*)d_in[1];
  const float* xref  = (const float*)d_in[2];
  const float* dtok  = (const float*)d_in[3];
  const float* stok  = (const float*)d_in[4];
  const float* dmu   = (const float*)d_in[5];
  const float* dls   = (const float*)d_in[6];
  const float* rpbt  = (const float*)d_in[7];
  const float* qkvw  = (const float*)d_in[8];
  const float* qkvb  = (const float*)d_in[9];
  const float* rqkw  = (const float*)d_in[10];
  const float* rqkb  = (const float*)d_in[11];
  const float* convw = (const float*)d_in[12];
  const float* convb = (const float*)d_in[13];
  const float* projw = (const float*)d_in[14];
  const float* projb = (const float*)d_in[15];
  float* out = (float*)d_out;

  char* ws = (char*)d_ws;
  size_t off = 0;
  auto alloc = [&](size_t bytes) {
    char* p = ws + off;
    off += (bytes + 255) & ~(size_t)255;
    return p;
  };
  _Float16* q16  = (_Float16*)alloc((size_t)BQ * HH * NT * HD * 2);
  _Float16* k16  = (_Float16*)alloc((size_t)BQ * HH * NT * HD * 2);
  _Float16* vh16 = (_Float16*)alloc((size_t)BQ * HH * NT * HD * 2);
  _Float16* rq16 = (_Float16*)alloc((size_t)RB * HH * NRF * HD * 2);
  _Float16* rv16 = (_Float16*)alloc((size_t)RB * HH * NRF * HD * 2);
  float* rbuf  = (float*)alloc((size_t)RB * HH * SPAT * CONVW * 4);
  float* ubuf  = (float*)alloc((size_t)RB * HH * SPAT * CONVW * 4);
  float* stats = (float*)alloc(64 * 2 * 4);
  // aliases (regions are dead by the time these are written):
  _Float16* qn16   = (_Float16*)ubuf;  // q_new, written after last LN/GELU use of u
  _Float16* outpre = (_Float16*)rbuf;  // pre-projection output, after r consumed

  k_qkv<<<(1568 * 48) / 8, 256, 0, stream>>>(x, qkvw, qkvb, q16, k16, vh16);
  k_refqk<<<(32 * 32) / 8, 256, 0, stream>>>(xref, rqkw, rqkb, dmu, dls, rq16, rv16);
  k_refattn<<<(BQ * HH * 16) / 8, 256, 0, stream>>>(q16, rq16, rbuf);
  for (int it = 0; it < 3; ++it) {
    k_conv<<<(RB * HH * SPAT * CONVW) / 256, 256, 0, stream>>>(rbuf, convw, convb, ubuf);
    k_stats<<<RB * HH, 256, 0, stream>>>(ubuf, stats);
    k_gelu<<<(RB * HH * SPAT * CONVW) / 256, 256, 0, stream>>>(rbuf, ubuf, stats);
  }
  k_refsm_qnew<<<BQ * HH, 256, 0, stream>>>(rbuf, rv16, qn16);
  k_winattn<<<BQ * HH, 256, 0, stream>>>(qn16, k16, vh16, rpbt, mask, outpre);
  k_proj<<<(1568 * 16) / 8, 256, 0, stream>>>(outpre, projw, projb, out);
  k_tokens<<<16, 256, 0, stream>>>(dtok, stok, out);
}
